// Net_55130200211993
// MI455X (gfx1250) — matrix operand order repaired and checked by reading, not compiled, not run on hardware
//
#include <hip/hip_runtime.h>

// ============================================================================
// SR3 UNet encoder forward on MI455X (gfx1250, wave32, WMMA).
//
// Roofline: ~15 GFLOP total; dominated by the 96x96 self-attention (8.2 GFLOP)
// and the 3x3 convs. Working set (~13 MB incl. f16 packed weights) lives in
// the 192 MB L2 -> matrix-core / L2 bound, HBM irrelevant. The 9216x9216
// attention map (340 MB) is never materialized: flash-style streaming softmax.
//
// All GEMMs use v_wmma_f32_16x16x32_f16. Every fragment is one (or two)
// aligned global_load_b128:
//   * weights pre-packed into per-lane A-fragment order (f16)
//   * activations stored f16 transposed (pixel-major, Cpad32 channel stride,
//     zero-filled pads); conv halo handled by an address-select to a zeroed
//     row -> no per-element masking in the hot loop
//   * every conv wave owns TWO spatial tiles: each A fragment feeds 2 wmma
//   * attention: q/k transposed f16 (pad-zeroed), v row-major f16; Q staged in
//     LDS pre-swizzled (ds_load_b128 in the hot loop); 2 query tiles per wave
//     share every K/V fragment (4 wmma per K-fragment pair).
// ============================================================================

typedef __attribute__((ext_vector_type(16))) _Float16 v16h;
typedef __attribute__((ext_vector_type(8)))  _Float16 v8h;
typedef __attribute__((ext_vector_type(8)))  float    v8f;

// K-index map for the 16-bit A matrix (16x32, MxK), contiguous order used by this code (B loads match):
//   K = 16*half + e
__device__ __forceinline__ int a_kmap(int e, int half) {
    return e + 16 * half;
}

__device__ __forceinline__ v8f wmma_f16(v16h a, v16h b, v8f c) {
    return __builtin_amdgcn_wmma_f32_16x16x32_f16(false, a, false, b,
                                                  (short)0, c, false, false);
}

// ---------------------------------------------------------------------------
// Pack f32 weights into f16 A-fragment-linear layout:
//   wp[((mt*KC + kc)*32 + lane)*16 + e] = W[M=mt*16+(lane&15)][k=kc*32+a_kmap]
// 3x3 convs: K reordered (kh,kw)-major / channel-minor, channels padded to 32.
// ---------------------------------------------------------------------------
__global__ void k_pack_wfrag(const float* __restrict__ w, _Float16* __restrict__ wp,
                             int Cout, int Cin, int is3x3) {
    const int Cpin = (Cin + 31) & ~31;
    const int Kpad = is3x3 ? 9 * Cpin : Cpin;
    const int KC = Kpad >> 5;
    const int Mt = (Cout + 15) >> 4;
    const int total = Mt * KC * 512;
    int idx = blockIdx.x * 256 + threadIdx.x;
    if (idx >= total) return;
    int e = idx & 15, lane = (idx >> 4) & 31, t = idx >> 9;
    int kc = t % KC, mt = t / KC;
    int half = lane >> 4, l15 = lane & 15;
    int M = mt * 16 + l15;
    int k = kc * 32 + a_kmap(e, half);
    float v = 0.f;
    if (M < Cout) {
        if (is3x3) {
            int r9 = k / Cpin, ci = k - r9 * Cpin;
            if (ci < Cin) v = w[(M * Cin + ci) * 9 + r9];   // w: (Cout,Cin,3,3)
        } else if (k < Cin) {
            v = w[M * Cin + k];
        }
    }
    wp[idx] = (_Float16)v;
}

// Zero an f16 range (used for the halo zero-row).
__global__ void k_zero16(_Float16* __restrict__ p, int count) {
    int i = blockIdx.x * 256 + threadIdx.x;
    if (i < count) p[i] = (_Float16)0.f;
}

// ---------------------------------------------------------------------------
// 3x3 conv (pad=1, stride 1|2), implicit-GEMM WMMA. One wave = 16 out-chan x
// 32 spatial (two 16-wide tiles sharing each A fragment). Halo -> zero-row
// address select (no per-element masking). Nout is a multiple of 32.
// ---------------------------------------------------------------------------
__global__ void k_conv3x3_wmma(const _Float16* __restrict__ xT,
                               const _Float16* __restrict__ zrow,
                               const _Float16* __restrict__ wp,
                               const float* __restrict__ bias,
                               const float* __restrict__ sAff,
                               const float* __restrict__ res,
                               float* __restrict__ out32,
                               _Float16* __restrict__ out16T,
                               int Cin, int Cout, int Hin, int Win, int stride) {
    const int lane = threadIdx.x;
    const int half = lane >> 4, l15 = lane & 15;
    const int Cpin = (Cin + 31) & ~31;
    const int Hout = (Hin - 1) / stride + 1;
    const int Wout = (Win - 1) / stride + 1;
    const int Nout = Hout * Wout;
    const int m0 = blockIdx.x * 16;
    const int n0 = blockIdx.y * 32;

    const int nA = n0 + l15, nB = n0 + 16 + l15;
    const int hoA = nA / Wout, woA = nA - hoA * Wout;
    const int hoB = nB / Wout, woB = nB - hoB * Wout;

    const int KC = (9 * Cpin) >> 5;
    const _Float16* wl = wp + (size_t)blockIdx.x * KC * 512 + (size_t)lane * 16;

    v8f accA = {}, accB = {};
    int kc = 0;
#pragma unroll
    for (int r9 = 0; r9 < 9; ++r9) {
        const int kh = r9 / 3, kw = r9 % 3;          // compile-time constants
        const int hiA = hoA * stride + kh - 1, wiA = woA * stride + kw - 1;
        const int hiB = hoB * stride + kh - 1, wiB = woB * stride + kw - 1;
        const bool okA = (hiA >= 0) && (hiA < Hin) && (wiA >= 0) && (wiA < Win);
        const bool okB = (hiB >= 0) && (hiB < Hin) && (wiB >= 0) && (wiB < Win);
        const _Float16* pA = okA ? (xT + (size_t)(hiA * Win + wiA) * Cpin) : zrow;
        const _Float16* pB = okB ? (xT + (size_t)(hiB * Win + wiB) * Cpin) : zrow;
        for (int cc = 0; cc < Cpin; cc += 32, ++kc) {
            v16h a  = *(const v16h*)(wl + (size_t)kc * 512);
            v16h bA = *(const v16h*)(pA + cc + 16 * half);
            v16h bB = *(const v16h*)(pB + cc + 16 * half);
            accA = wmma_f16(a, bA, accA);
            accB = wmma_f16(a, bB, accB);
        }
    }
    const int Cpout = (Cout + 31) & ~31;
#pragma unroll
    for (int r = 0; r < 8; ++r) {
        const int c = m0 + r + 8 * half;             // D layout: row = r+8*half
        const float bs = (c < Cout) ? bias[c] : 0.f;
        float vA = accA[r] + bs, vB = accB[r] + bs;
        if (sAff && c < Cout) {
            float s = sAff[c];
            vA = vA + s * vA + s; vB = vB + s * vB + s;
        }
        if (res && c < Cout) {
            vA += res[(size_t)c * Nout + nA];
            vB += res[(size_t)c * Nout + nB];
        }
        if (out32 && c < Cout) {
            out32[(size_t)c * Nout + nA] = vA;
            out32[(size_t)c * Nout + nB] = vB;
        }
        if (out16T && c < Cpout) {
            out16T[(size_t)nA * Cpout + c] = (c < Cout) ? (_Float16)vA : (_Float16)0.f;
            out16T[(size_t)nB * Cpout + c] = (c < Cout) ? (_Float16)vB : (_Float16)0.f;
        }
    }
}

// ---------------------------------------------------------------------------
// 1x1 conv GEMM, two spatial tiles per wave (N multiple of 32).
// ---------------------------------------------------------------------------
__global__ void k_conv1x1_wmma(const _Float16* __restrict__ xT,   // N x Cpin
                               const _Float16* __restrict__ wp,
                               const float* __restrict__ bias,
                               const float* __restrict__ res,
                               float* __restrict__ out32,
                               _Float16* __restrict__ out16T,
                               int Cin, int Cout, int N) {
    const int lane = threadIdx.x;
    const int half = lane >> 4, l15 = lane & 15;
    const int Cpin = (Cin + 31) & ~31;
    const int KC = Cpin >> 5;
    const int m0 = blockIdx.x * 16;
    const int nA = blockIdx.y * 32 + l15, nB = nA + 16;

    const _Float16* wl  = wp + (size_t)blockIdx.x * KC * 512 + (size_t)lane * 16;
    const _Float16* xlA = xT + (size_t)nA * Cpin + 16 * half;
    const _Float16* xlB = xT + (size_t)nB * Cpin + 16 * half;

    v8f accA = {}, accB = {};
    for (int kc = 0; kc < KC; ++kc) {
        v16h a  = *(const v16h*)(wl + (size_t)kc * 512);
        v16h bA = *(const v16h*)(xlA + kc * 32);
        v16h bB = *(const v16h*)(xlB + kc * 32);
        accA = wmma_f16(a, bA, accA);
        accB = wmma_f16(a, bB, accB);
    }
    const int Cpout = (Cout + 31) & ~31;
#pragma unroll
    for (int r = 0; r < 8; ++r) {
        const int c = m0 + r + 8 * half;
        const float bs = (bias && c < Cout) ? bias[c] : 0.f;
        float vA = accA[r] + bs, vB = accB[r] + bs;
        if (res && c < Cout) {
            vA += res[(size_t)c * N + nA];
            vB += res[(size_t)c * N + nB];
        }
        if (out32 && c < Cout) {
            out32[(size_t)c * N + nA] = vA;
            out32[(size_t)c * N + nB] = vB;
        }
        if (out16T && c < Cpout) {
            out16T[(size_t)nA * Cpout + c] = (c < Cout) ? (_Float16)vA : (_Float16)0.f;
            out16T[(size_t)nB * Cpout + c] = (c < Cout) ? (_Float16)vB : (_Float16)0.f;
        }
    }
}

// ---------------------------------------------------------------------------
// qkv 1x1 conv (no bias): scatters into qT/kT (N x Cq transposed f16) and
// vH (C x N row-major f16). Cout = 3C; two spatial tiles per wave.
// ---------------------------------------------------------------------------
__global__ void k_qkv_wmma(const _Float16* __restrict__ xT,
                           const _Float16* __restrict__ wp,
                           _Float16* __restrict__ qT, _Float16* __restrict__ kT,
                           _Float16* __restrict__ vH,
                           int Cin, int C, int N) {
    const int lane = threadIdx.x;
    const int half = lane >> 4, l15 = lane & 15;
    const int Cpin = (Cin + 31) & ~31;
    const int KC = Cpin >> 5;
    const int m0 = blockIdx.x * 16;
    const int nA = blockIdx.y * 32 + l15, nB = nA + 16;
    const int Cq = (C + 31) & ~31;

    const _Float16* wl  = wp + (size_t)blockIdx.x * KC * 512 + (size_t)lane * 16;
    const _Float16* xlA = xT + (size_t)nA * Cpin + 16 * half;
    const _Float16* xlB = xT + (size_t)nB * Cpin + 16 * half;

    v8f accA = {}, accB = {};
    for (int kc = 0; kc < KC; ++kc) {
        v16h a  = *(const v16h*)(wl + (size_t)kc * 512);
        v16h bA = *(const v16h*)(xlA + kc * 32);
        v16h bB = *(const v16h*)(xlB + kc * 32);
        accA = wmma_f16(a, bA, accA);
        accB = wmma_f16(a, bB, accB);
    }
#pragma unroll
    for (int r = 0; r < 8; ++r) {
        const int c = m0 + r + 8 * half;
        _Float16 vA = (_Float16)accA[r], vB = (_Float16)accB[r];
        if (c < C) {
            qT[(size_t)nA * Cq + c] = vA;
            qT[(size_t)nB * Cq + c] = vB;
        } else if (c < 2 * C) {
            kT[(size_t)nA * Cq + (c - C)] = vA;
            kT[(size_t)nB * Cq + (c - C)] = vB;
        } else if (c < 3 * C) {
            vH[(size_t)(c - 2 * C) * N + nA] = vA;
            vH[(size_t)(c - 2 * C) * N + nB] = vB;
        }
    }
}

// Zero the pad channels [C, Cq) of a transposed N x Cq f16 buffer.
__global__ void k_pad_zero(_Float16* __restrict__ p, int C, int Cq, int N) {
    const int pad = Cq - C;
    int i = blockIdx.x * 256 + threadIdx.x;
    if (i < N * pad) {
        int nn = i / pad, c = C + (i - nn * pad);
        p[(size_t)nn * Cq + c] = (_Float16)0.f;
    }
}

// Convert the 1-channel f32 input to f16 transposed (Cpin=32, pads zero).
__global__ void k_cvt_in(const float* __restrict__ x, _Float16* __restrict__ xT, int N) {
    int i = blockIdx.x * 256 + threadIdx.x;
    if (i < N * 32) {
        int nn = i >> 5, c = i & 31;
        xT[i] = (c == 0) ? (_Float16)x[nn] : (_Float16)0.f;
    }
}

// ---------------------------------------------------------------------------
// GroupNorm (+ optional Swish). f32 (C,N) in, f16 transposed (N,Cp32) out with
// zero-filled pad channels. One block per group; float4 stats pass.
// ---------------------------------------------------------------------------
__global__ void k_groupnorm(const float* __restrict__ x, _Float16* __restrict__ outT,
                            const float* __restrict__ gamma,
                            const float* __restrict__ beta,
                            int C, int N, int groups, int do_swish) {
    const int gi  = blockIdx.x;
    const int cpg = C / groups;
    const int cnt = cpg * N;
    const float* xg = x + (size_t)gi * cpg * N;

    float s = 0.f, s2 = 0.f;
    const float4* x4 = (const float4*)xg;
    for (int i = threadIdx.x; i < (cnt >> 2); i += blockDim.x) {
        float4 v = x4[i];
        s  += v.x + v.y + v.z + v.w;
        s2 += v.x * v.x + v.y * v.y + v.z * v.z + v.w * v.w;
    }
    __shared__ float red[512];
    red[threadIdx.x] = s; red[256 + threadIdx.x] = s2;
    __syncthreads();
    for (int off = 128; off > 0; off >>= 1) {
        if (threadIdx.x < off) {
            red[threadIdx.x]       += red[threadIdx.x + off];
            red[256 + threadIdx.x] += red[256 + threadIdx.x + off];
        }
        __syncthreads();
    }
    const float mean = red[0] / (float)cnt;
    const float var  = red[256] / (float)cnt - mean * mean;
    const float rstd = rsqrtf(var + 1e-5f);

    const int Cp = (C + 31) & ~31;
    for (int i = threadIdx.x; i < cnt; i += blockDim.x) {
        int cl = i / N, nn = i - cl * N;
        int c  = gi * cpg + cl;
        float v = (xg[i] - mean) * rstd * gamma[c] + beta[c];
        if (do_swish) v = v / (1.f + __expf(-v));
        outT[(size_t)nn * Cp + c] = (_Float16)v;
    }
    if (gi == 0 && Cp > C) {                 // zero pad channels once
        int pad = Cp - C;
        for (int i = threadIdx.x; i < pad * N; i += blockDim.x) {
            int nn = i / pad, c = C + (i - nn * pad);
            outT[(size_t)nn * Cp + c] = (_Float16)0.f;
        }
    }
}

// ---------------------------------------------------------------------------
// Flash attention, f16 in/out. One wave per block owns TWO 16-query tiles and
// 48 channels; streams 32 keys/iter. All N are multiples of 32 -> no key
// masking. K/V fragments shared across both query tiles (4 wmma per K pair).
// ---------------------------------------------------------------------------
__device__ __forceinline__ void softmax_update(v8f& s0, v8f& s1, float scale,
                                               float* rmax, float* rsum,
                                               v8f& o0, v8f& o1, v8f& o2,
                                               _Float16* ldsPq, int half, int l15) {
#pragma unroll
    for (int r = 0; r < 8; ++r) {
        float v0 = s0[r] * scale, v1 = s1[r] * scale;
        float m = fmaxf(v0, v1);
#pragma unroll
        for (int o = 1; o < 16; o <<= 1) m = fmaxf(m, __shfl_xor(m, o, 32));
        float nm   = fmaxf(rmax[r], m);
        float corr = __expf(rmax[r] - nm);
        rmax[r] = nm;
        float e0 = __expf(v0 - nm), e1 = __expf(v1 - nm);
        float ss = e0 + e1;
#pragma unroll
        for (int o = 1; o < 16; o <<= 1) ss += __shfl_xor(ss, o, 32);
        rsum[r] = rsum[r] * corr + ss;
        o0[r] *= corr; o1[r] *= corr; o2[r] *= corr;
        int row = r + 8 * half;                        // D layout -> P transpose
        ldsPq[row * 32 + l15]      = (_Float16)e0;
        ldsPq[row * 32 + 16 + l15] = (_Float16)e1;
    }
}

__global__ void k_attn_flash_wmma(const _Float16* __restrict__ qT,
                                  const _Float16* __restrict__ kT,
                                  const _Float16* __restrict__ vH,
                                  _Float16* __restrict__ outT,
                                  int C, int N) {
    const int lane = threadIdx.x;
    const int half = lane >> 4, l15 = lane & 15;
    const int Cq  = (C + 31) & ~31;
    const int nkc = Cq >> 5;
    const int m0  = blockIdx.x * 32;
    const int c0  = blockIdx.y * 48;
    const float scale = rsqrtf((float)C);

    __shared__ _Float16 ldsQ[2 * 12 * 512];   // fragment-order Q, <=24 KB
    __shared__ _Float16 ldsP[2 * 512];        // P transpose, 2 KB

    for (int qt = 0; qt < 2; ++qt) {          // stage Q pre-swizzled
        const _Float16* src = qT + (size_t)(m0 + qt * 16 + l15) * Cq;
        for (int kc = 0; kc < nkc; ++kc) {
            v8h lo  = *(const v8h*)(src + kc * 32 + 16 * half);
            v8h hi8 = *(const v8h*)(src + kc * 32 + 16 * half + 8);
            v16h tmp;
#pragma unroll
            for (int e = 0; e < 8; ++e) { tmp[e] = lo[e]; tmp[8 + e] = hi8[e]; }
            *(v16h*)&ldsQ[((qt * nkc + kc) * 32 + lane) * 16] = tmp;
        }
    }
    __syncthreads();

    v8f o00 = {}, o01 = {}, o02 = {}, o10 = {}, o11 = {}, o12 = {};
    float rmax0[8], rsum0[8], rmax1[8], rsum1[8];
#pragma unroll
    for (int r = 0; r < 8; ++r) {
        rmax0[r] = -1e30f; rsum0[r] = 0.f; rmax1[r] = -1e30f; rsum1[r] = 0.f;
    }

    for (int y0 = 0; y0 < N; y0 += 32) {
        v8f s00 = {}, s01 = {}, s10 = {}, s11 = {};
        const _Float16* kp0 = kT + (size_t)(y0 + l15) * Cq + 16 * half;
        const _Float16* kp1 = kT + (size_t)(y0 + 16 + l15) * Cq + 16 * half;
        for (int kc = 0; kc < nkc; ++kc) {
            v16h b0 = *(const v16h*)(kp0 + kc * 32);
            v16h b1 = *(const v16h*)(kp1 + kc * 32);
            v16h a0 = *(const v16h*)&ldsQ[(kc * 32 + lane) * 16];
            v16h a1 = *(const v16h*)&ldsQ[((nkc + kc) * 32 + lane) * 16];
            s00 = wmma_f16(a0, b0, s00); s01 = wmma_f16(a0, b1, s01);
            s10 = wmma_f16(a1, b0, s10); s11 = wmma_f16(a1, b1, s11);
        }

        softmax_update(s00, s01, scale, rmax0, rsum0, o00, o01, o02, ldsP,       half, l15);
        softmax_update(s10, s11, scale, rmax1, rsum1, o10, o11, o12, ldsP + 512, half, l15);
        __syncthreads();

        v16h ap0, ap1;
        {
            const _Float16* p0 = ldsP + l15 * 32;
            const _Float16* p1 = ldsP + 512 + l15 * 32;
            v8h a = *(const v8h*)(p0 + 16 * half), b = *(const v8h*)(p0 + 16 * half + 8);
            v8h c = *(const v8h*)(p1 + 16 * half), d = *(const v8h*)(p1 + 16 * half + 8);
#pragma unroll
            for (int e = 0; e < 8; ++e) {
                ap0[e] = a[e]; ap0[8 + e] = b[e];
                ap1[e] = c[e]; ap1[8 + e] = d[e];
            }
        }

#pragma unroll
        for (int t = 0; t < 3; ++t) {                 // V fragments shared by both qtiles
            int ch  = c0 + t * 16 + l15;
            int chc = (ch < C) ? ch : (C - 1);
            v16h bv = *(const v16h*)(vH + (size_t)chc * N + y0 + 16 * half);
            v16h z = {};
            bv = (ch < C) ? bv : z;
            if (t == 0) { o00 = wmma_f16(ap0, bv, o00); o10 = wmma_f16(ap1, bv, o10); }
            if (t == 1) { o01 = wmma_f16(ap0, bv, o01); o11 = wmma_f16(ap1, bv, o11); }
            if (t == 2) { o02 = wmma_f16(ap0, bv, o02); o12 = wmma_f16(ap1, bv, o12); }
        }
        __syncthreads();
    }

    const int Cpo = (C + 31) & ~31;                   // out: N x Cpo transposed f16
#pragma unroll
    for (int t = 0; t < 3; ++t) {
#pragma unroll
        for (int r = 0; r < 8; ++r) {
            int ch = c0 + t * 16 + l15;
            int mA = m0 + r + 8 * half;
            int mB = m0 + 16 + r + 8 * half;
            float vA = (t == 0 ? o00[r] : t == 1 ? o01[r] : o02[r]) / rsum0[r];
            float vB = (t == 0 ? o10[r] : t == 1 ? o11[r] : o12[r]) / rsum1[r];
            if (ch < Cpo) {
                outT[(size_t)mA * Cpo + ch] = (ch < C) ? (_Float16)vA : (_Float16)0.f;
                outT[(size_t)mB * Cpo + ch] = (ch < C) ? (_Float16)vB : (_Float16)0.f;
            }
        }
    }
}

// ---------------------------------------------------------------------------
// Noise MLP + all 5 featurewise-affine s vectors (tiny, 1 block).
// ---------------------------------------------------------------------------
struct AffineParams {
    const float *w1, *b1, *w2, *b2;
    const float *fw_w[5], *fw_b[5];
    float *s_out[5];
    int dims[5];
};

__global__ void k_noise_affine(const float* __restrict__ noise, AffineParams P) {
    __shared__ float enc[24], hmid[96], tvec[24];
    const int tid = threadIdx.x;
    if (tid < 12) {
        float step = (float)tid / 12.f;
        float e = noise[0] * __expf(-logf(10000.f) * step);
        enc[tid]      = sinf(e);
        enc[tid + 12] = cosf(e);
    }
    __syncthreads();
    if (tid < 96) {
        float a = P.b1[tid];
        for (int i = 0; i < 24; ++i) a += enc[i] * P.w1[i * 96 + tid];
        hmid[tid] = a / (1.f + __expf(-a));
    }
    __syncthreads();
    if (tid < 24) {
        float a = P.b2[tid];
        for (int j = 0; j < 96; ++j) a += hmid[j] * P.w2[j * 24 + tid];
        tvec[tid] = a;
    }
    __syncthreads();
    for (int b = 0; b < 5; ++b) {
        int D = P.dims[b];
        for (int c = tid; c < D; c += blockDim.x) {
            float g = P.fw_b[b][c] + P.fw_b[b][D + c];
            for (int i = 0; i < 24; ++i)
                g += tvec[i] * (P.fw_w[b][i * 2 * D + c] + P.fw_w[b][i * 2 * D + D + c]);
            P.s_out[b][c] = 0.5f * g;
        }
    }
}

// ===========================================================================
extern "C" void kernel_launch(void* const* d_in, const int* in_sizes, int n_in,
                              void* d_out, int out_size, void* d_ws, size_t ws_size,
                              hipStream_t stream) {
    (void)in_sizes; (void)n_in; (void)out_size; (void)ws_size;
    auto IN = [&](int i) { return (const float*)d_in[i]; };

    float* ws = (float*)d_ws;
    float* sA1 = ws + 0;   float* sR1 = ws + 64;  float* sA2 = ws + 128;
    float* sR2 = ws + 256; float* sA3 = ws + 384;                // within [0,1024)
    const size_t SZ = 221184;                                     // 384*576 == 24*9216
    float* bufA = ws + 1024;
    float* bufC = bufA + SZ;
    float* bufD = bufC + SZ;
    float* bufE = bufD + SZ;

    _Float16* H = (_Float16*)(bufE + SZ);                         // 32B-aligned region
    size_t ho16 = 0;
    auto alloc16 = [&](size_t n) { _Float16* p = H + ho16; ho16 += (n + 15) & ~(size_t)15; return p; };
    _Float16* zrow  = alloc16(384);                               // halo zero-row
    _Float16* xh    = alloc16(9216 * 32);
    _Float16* bufBh = alloc16(294912);
    _Float16* bufCh = alloc16(294912);
    _Float16* bufDh = alloc16(294912);
    _Float16* bufAh = alloc16(73728);
    _Float16* qTb   = alloc16(294912 + 64);
    _Float16* kTb   = alloc16(294912 + 64);
    _Float16* vHb   = alloc16(221184);

    auto pack = [&](int widx, int Cout, int Cin, int is3x3) -> const _Float16* {
        int Cpin = (Cin + 31) & ~31;
        int KC = (is3x3 ? 9 * Cpin : Cpin) >> 5;
        int Mt = (Cout + 15) >> 4;
        size_t total = (size_t)Mt * KC * 512;
        _Float16* dst = (_Float16*)alloc16(total);
        k_pack_wfrag<<<(int)((total + 255) / 256), 256, 0, stream>>>(IN(widx), dst, Cout, Cin, is3x3);
        return dst;
    };
    auto conv3 = [&](const _Float16* x, const _Float16* wp, const float* b,
                     const float* s, const float* res, float* o32, _Float16* o16,
                     int Cin, int Cout, int Hin, int Win, int stride) {
        int Hout = (Hin - 1) / stride + 1, Wout = (Win - 1) / stride + 1;
        dim3 g((Cout + 15) / 16, (Hout * Wout) / 32);
        k_conv3x3_wmma<<<g, 32, 0, stream>>>(x, zrow, wp, b, s, res, o32, o16,
                                             Cin, Cout, Hin, Win, stride);
    };
    auto conv1 = [&](const _Float16* x, const _Float16* wp, const float* b,
                     const float* res, float* o32, _Float16* o16,
                     int Cin, int Cout, int N) {
        dim3 g((Cout + 15) / 16, N / 32);
        k_conv1x1_wmma<<<g, 32, 0, stream>>>(x, wp, b, res, o32, o16, Cin, Cout, N);
    };
    auto qkv = [&](const _Float16* x, const _Float16* wp, int Cin, int C, int N) {
        dim3 g((3 * C + 15) / 16, N / 32);
        k_qkv_wmma<<<g, 32, 0, stream>>>(x, wp, qTb, kTb, vHb, Cin, C, N);
        int Cq = (C + 31) & ~31;
        if (Cq > C) {
            int cnt = N * (Cq - C);
            k_pad_zero<<<(cnt + 255) / 256, 256, 0, stream>>>(qTb, C, Cq, N);
            k_pad_zero<<<(cnt + 255) / 256, 256, 0, stream>>>(kTb, C, Cq, N);
        }
    };
    auto gn = [&](const float* x, _Float16* o, const float* ga, const float* be,
                  int C, int N, int grp, int sw) {
        k_groupnorm<<<grp, 256, 0, stream>>>(x, o, ga, be, C, N, grp, sw);
    };
    auto attn = [&](_Float16* o, int C, int N) {
        dim3 g(N / 32, (C + 47) / 48);
        k_attn_flash_wmma<<<g, 32, 0, stream>>>(qTb, kTb, vHb, o, C, N);
    };

    // ---- noise MLP + affine s vectors; input convert; zero-row ----
    AffineParams AP;
    AP.w1 = IN(4); AP.b1 = IN(5); AP.w2 = IN(6); AP.b2 = IN(7);
    AP.fw_w[0] = IN(12); AP.fw_b[0] = IN(13); AP.s_out[0] = sA1; AP.dims[0] = 24;
    AP.fw_w[1] = IN(27); AP.fw_b[1] = IN(28); AP.s_out[1] = sR1; AP.dims[1] = 24;
    AP.fw_w[2] = IN(39); AP.fw_b[2] = IN(40); AP.s_out[2] = sA2; AP.dims[2] = 96;
    AP.fw_w[3] = IN(56); AP.fw_b[3] = IN(57); AP.s_out[3] = sR2; AP.dims[3] = 96;
    AP.fw_w[4] = IN(68); AP.fw_b[4] = IN(69); AP.s_out[4] = sA3; AP.dims[4] = 384;
    k_noise_affine<<<1, 384, 0, stream>>>(IN(1), AP);
    k_zero16<<<2, 256, 0, stream>>>(zrow, 384);
    k_cvt_in<<<(9216 * 32 + 255) / 256, 256, 0, stream>>>(IN(0), xh, 9216);

    // ---- pack all weights (f16 fragment layout) ----
    const _Float16 *wp_c0   = pack(2, 24, 1, 1);
    const _Float16 *wp_a1c1 = pack(10, 24, 24, 1), *wp_a1c2 = pack(16, 24, 24, 1);
    const _Float16 *wp_a1qk = pack(20, 72, 24, 0), *wp_a1ao = pack(21, 24, 24, 0);
    const _Float16 *wp_r1c1 = pack(25, 24, 24, 1), *wp_r1c2 = pack(31, 24, 24, 1);
    const _Float16 *wp_d1   = pack(33, 24, 24, 1);
    const _Float16 *wp_a2c1 = pack(37, 96, 24, 1), *wp_a2c2 = pack(43, 96, 96, 1);
    const _Float16 *wp_a2rc = pack(45, 96, 24, 0);
    const _Float16 *wp_a2qk = pack(49, 288, 96, 0), *wp_a2ao = pack(50, 96, 96, 0);
    const _Float16 *wp_r2c1 = pack(54, 96, 96, 1), *wp_r2c2 = pack(60, 96, 96, 1);
    const _Float16 *wp_d2   = pack(62, 96, 96, 1);
    const _Float16 *wp_a3c1 = pack(66, 384, 96, 1), *wp_a3c2 = pack(72, 384, 384, 1);
    const _Float16 *wp_a3rc = pack(74, 384, 96, 0);
    const _Float16 *wp_a3qk = pack(78, 1152, 384, 0), *wp_a3ao = pack(79, 384, 384, 0);

    // ---- stage 1: 96x96, C=24, N=9216 ----
    conv3(xh, wp_c0, IN(3), nullptr, nullptr, bufA, nullptr, 1, 24, 96, 96, 1);
    gn(bufA, bufBh, IN(8), IN(9), 24, 9216, 3, 1);
    conv3(bufBh, wp_a1c1, IN(11), sA1, nullptr, bufC, nullptr, 24, 24, 96, 96, 1);
    gn(bufC, bufBh, IN(14), IN(15), 24, 9216, 3, 1);
    conv3(bufBh, wp_a1c2, IN(17), nullptr, bufA, bufD, nullptr, 24, 24, 96, 96, 1);
    gn(bufD, bufBh, IN(18), IN(19), 24, 9216, 3, 0);
    qkv(bufBh, wp_a1qk, 24, 24, 9216);
    attn(bufCh, 24, 9216);
    conv1(bufCh, wp_a1ao, IN(22), bufD, bufA, nullptr, 24, 24, 9216);
    gn(bufA, bufBh, IN(23), IN(24), 24, 9216, 3, 1);
    conv3(bufBh, wp_r1c1, IN(26), sR1, nullptr, bufC, nullptr, 24, 24, 96, 96, 1);
    gn(bufC, bufBh, IN(29), IN(30), 24, 9216, 3, 1);
    conv3(bufBh, wp_r1c2, IN(32), nullptr, bufA, nullptr, bufDh, 24, 24, 96, 96, 1);
    conv3(bufDh, wp_d1, IN(34), nullptr, nullptr, bufA, bufAh, 24, 24, 96, 96, 2);

    // ---- stage 2: 48x48, N=2304 ----
    gn(bufA, bufBh, IN(35), IN(36), 24, 2304, 3, 1);
    conv3(bufBh, wp_a2c1, IN(38), sA2, nullptr, bufC, nullptr, 24, 96, 48, 48, 1);
    gn(bufC, bufBh, IN(41), IN(42), 96, 2304, 3, 1);
    conv1(bufAh, wp_a2rc, IN(46), nullptr, bufE, nullptr, 24, 96, 2304);
    conv3(bufBh, wp_a2c2, IN(44), nullptr, bufE, bufD, nullptr, 96, 96, 48, 48, 1);
    gn(bufD, bufBh, IN(47), IN(48), 96, 2304, 3, 0);
    qkv(bufBh, wp_a2qk, 96, 96, 2304);
    attn(bufCh, 96, 2304);
    conv1(bufCh, wp_a2ao, IN(51), bufD, bufA, nullptr, 96, 96, 2304);
    gn(bufA, bufBh, IN(52), IN(53), 96, 2304, 12, 1);
    conv3(bufBh, wp_r2c1, IN(55), sR2, nullptr, bufC, nullptr, 96, 96, 48, 48, 1);
    gn(bufC, bufBh, IN(58), IN(59), 96, 2304, 12, 1);
    conv3(bufBh, wp_r2c2, IN(61), nullptr, bufA, nullptr, bufDh, 96, 96, 48, 48, 1);
    conv3(bufDh, wp_d2, IN(63), nullptr, nullptr, bufA, bufAh, 96, 96, 48, 48, 2);

    // ---- stage 3: 24x24, N=576 ----
    gn(bufA, bufBh, IN(64), IN(65), 96, 576, 12, 1);
    conv3(bufBh, wp_a3c1, IN(67), sA3, nullptr, bufC, nullptr, 96, 384, 24, 24, 1);
    gn(bufC, bufBh, IN(70), IN(71), 384, 576, 12, 1);
    conv1(bufAh, wp_a3rc, IN(75), nullptr, bufE, nullptr, 96, 384, 576);
    conv3(bufBh, wp_a3c2, IN(73), nullptr, bufE, bufD, nullptr, 384, 384, 24, 24, 1);
    gn(bufD, bufBh, IN(76), IN(77), 384, 576, 12, 0);
    qkv(bufBh, wp_a3qk, 384, 384, 576);
    attn(bufCh, 384, 576);
    conv1(bufCh, wp_a3ao, IN(80), bufD, (float*)d_out, nullptr, 384, 384, 576);
}
